// NystromAttention_60679297958522
// MI455X (gfx1250) — matrix operand order repaired and checked by reading, not compiled, not run on hardware
//
#include <hip/hip_runtime.h>
#include <cstddef>

typedef __attribute__((ext_vector_type(16))) _Float16 v16h;
typedef __attribute__((ext_vector_type(8)))  _Float16 v8h;
typedef __attribute__((ext_vector_type(4)))  _Float16 v4h;
typedef __attribute__((ext_vector_type(8)))  float    v8f;

#define BH_TOTAL 2048
#define N_TOK 256
#define DHEAD 64
#define LMK 64

// LDS row strides (in halves); 72*2=144B and 264*2=528B are 16B-aligned rows
#define LDT  72
#define LD64 72
#define LD3  264

#define SZ_TOK (N_TOK * LDT * 2)
#define SZ_M64 (LMK * LD64 * 2)
#define SZ_T64 (LMK * LD3 * 2)
#define OFF_Q   0                       // q (f16) -> later kernel_1   [256][72]  A/B-use
#define OFF_K   (OFF_Q  + SZ_TOK)      // k (f16)                      [256][72]  B-use (load_Bt)
#define OFF_VT  (OFF_K  + SZ_TOK)      // v^T (f16)                    [64][264]  B-use
#define OFF_K3  (OFF_VT + SZ_T64)      // kernel_3                     [64][264]  A-use
#define OFF_QM  (OFF_K3 + SZ_T64)      // q_m                          [64][72]   A-use
#define OFF_KM  (OFF_QM + SZ_M64)      // k_m                          [64][72]   B-use
#define OFF_K2  (OFF_KM + SZ_M64)      // kernel_2                     [64][72]   A-use
#define OFF_VAN (OFF_K2 + SZ_M64)      // NS V ping (normal, A-use)
#define OFF_VAT (OFF_VAN + SZ_M64)     // NS V ping (transposed, B-use)
#define OFF_VBN (OFF_VAT + SZ_M64)     // NS V pong (normal)
#define OFF_VBT (OFF_VBN + SZ_M64)     // NS V pong (transposed)
#define OFF_KV  (OFF_VBT + SZ_M64)     // X = K2@V (normal, A-use); later Z^T
#define OFF_T1  (OFF_KV + SZ_M64)      // T1^T / T3^T; later W^T
#define OFF_T2  (OFF_T1 + SZ_M64)      // T2^T
#define OFF_RED (OFF_T2 + SZ_M64)
#define SMEM_BYTES (OFF_RED + 256)

// ---------------------------------------------------------------------------
// WMMA helpers (CDNA5 v_wmma_f32_16x16x32_f16, wave32)
// ---------------------------------------------------------------------------
__device__ __forceinline__ v8f wmma16(v16h a, v16h b, v8f c) {
  return __builtin_amdgcn_wmma_f32_16x16x32_f16(
      /*neg_a=*/false, a, /*neg_b=*/false, b,
      /*c_mod=*/(short)0, c, /*reuse_a=*/false, /*reuse_b=*/false);
}

// A fragment: 16x32 f16 tile at (m0,k0) of row-major matrix (ld in halves).
__device__ __forceinline__ v16h load_A(const _Float16* M, int ld, int m0, int k0, int lane) {
  const int r  = m0 + (lane & 15);
  const int kb = k0 + ((lane & 16) ? 16 : 0);
  v16h a;
#pragma unroll
  for (int e = 0; e < 8; ++e) a[e] = M[r * ld + kb + e];
#pragma unroll
  for (int e = 0; e < 8; ++e) a[8 + e] = M[r * ld + kb + 8 + e];
  return a;
}

// B fragment from S stored row-major [N][K] (i.e. S = B^T): B(k,n) = S[n][k].
// Contiguous 16-half read per lane -> 2x ds_load_b128.
__device__ __forceinline__ v16h load_Bt(const _Float16* S, int ld, int n0, int k0, int lane) {
  const int n  = n0 + (lane & 15);
  const int kb = k0 + ((lane & 16) ? 16 : 0);
  v16h b;
#pragma unroll
  for (int e = 0; e < 16; ++e) b[e] = S[n * ld + kb + e];
  return b;
}

// Normal-layout store (row-major D[m][n]); strided column stores per lane.
__device__ __forceinline__ void store_tile_h(_Float16* D, int ld, int m0, int n0, int lane,
                                             v8f acc, float diag, float scale) {
  const int n  = n0 + (lane & 15);
  const int mb = m0 + ((lane & 16) ? 8 : 0);
#pragma unroll
  for (int r = 0; r < 8; ++r) {
    const float dg = ((mb + r) == n) ? diag : 0.0f;
    D[(mb + r) * ld + n] = (_Float16)(dg + scale * acc[r]);
  }
}

// Transposed store: writes D^T[n][m]; 8 contiguous halves per lane -> ds_store_b128.
__device__ __forceinline__ void store_tile_hT(_Float16* Dt, int ld, int m0, int n0, int lane,
                                              v8f acc, float diag, float scale) {
  const int n  = n0 + (lane & 15);
  const int mb = m0 + ((lane & 16) ? 8 : 0);
  v8h t;
#pragma unroll
  for (int r = 0; r < 8; ++r) {
    const float dg = ((mb + r) == n) ? diag : 0.0f;
    t[r] = (_Float16)(dg + scale * acc[r]);
  }
  *reinterpret_cast<v8h*>(&Dt[n * ld + mb]) = t;
}

// 64x64x64: C = A @ B with A row-major and B supplied transposed (Bt = B^T).
// 16 tiles split 2 per wave. Optional normal and/or transposed destinations.
__device__ __forceinline__ void mm64(const _Float16* A, int lda,
                                     const _Float16* Bt, int ldbt,
                                     _Float16* Dn, float diagN, float scaleN,
                                     _Float16* Dt, float diagT, float scaleT,
                                     int wave, int lane) {
#pragma unroll
  for (int t = 0; t < 2; ++t) {
    const int tile = wave * 2 + t;
    const int mt = tile >> 2, nt = tile & 3;
    v8f acc = {};
#pragma unroll
    for (int k0 = 0; k0 < 64; k0 += 32) {
      v16h a = load_A(A, lda, mt * 16, k0, lane);
      v16h b = load_Bt(Bt, ldbt, nt * 16, k0, lane);
      acc = wmma16(a, b, acc);
    }
    if (Dn) store_tile_h (Dn, LD64, mt * 16, nt * 16, lane, acc, diagN, scaleN);
    if (Dt) store_tile_hT(Dt, LD64, mt * 16, nt * 16, lane, acc, diagT, scaleT);
  }
}

// Row softmax over NF fragments that together hold full rows (16-lane shfl).
template <int NF>
__device__ __forceinline__ void row_softmax(v8f* acc, int lane) {
#pragma unroll
  for (int r = 0; r < 8; ++r) {
    float m = acc[0][r];
#pragma unroll
    for (int f = 1; f < NF; ++f) m = fmaxf(m, acc[f][r]);
#pragma unroll
    for (int off = 1; off < 16; off <<= 1) m = fmaxf(m, __shfl_xor(m, off, 32));
    float s = 0.0f;
#pragma unroll
    for (int f = 0; f < NF; ++f) {
      const float e = __expf(acc[f][r] - m);
      acc[f][r] = e;
      s += e;
    }
#pragma unroll
    for (int off = 1; off < 16; off <<= 1) s += __shfl_xor(s, off, 32);
    const float inv = 1.0f / s;
#pragma unroll
    for (int f = 0; f < NF; ++f) acc[f][r] *= inv;
  }
}

// ---------------------------------------------------------------------------
// One workgroup per (b,h): fully fused Nystrom attention in LDS.
// ---------------------------------------------------------------------------
__global__ void __launch_bounds__(256)
nystrom_attn_kernel(const float* __restrict__ q, const float* __restrict__ k,
                    const float* __restrict__ v, float* __restrict__ out) {
  extern __shared__ char smem[];
  _Float16* Qh  = reinterpret_cast<_Float16*>(smem + OFF_Q);
  _Float16* Kh  = reinterpret_cast<_Float16*>(smem + OFF_K);
  _Float16* VhT = reinterpret_cast<_Float16*>(smem + OFF_VT);
  _Float16* K3h = reinterpret_cast<_Float16*>(smem + OFF_K3);
  _Float16* QmH = reinterpret_cast<_Float16*>(smem + OFF_QM);
  _Float16* KmH = reinterpret_cast<_Float16*>(smem + OFF_KM);
  _Float16* K2h = reinterpret_cast<_Float16*>(smem + OFF_K2);
  _Float16* VaN = reinterpret_cast<_Float16*>(smem + OFF_VAN);
  _Float16* VaT = reinterpret_cast<_Float16*>(smem + OFF_VAT);
  _Float16* VbN = reinterpret_cast<_Float16*>(smem + OFF_VBN);
  _Float16* VbT = reinterpret_cast<_Float16*>(smem + OFF_VBT);
  _Float16* KVh = reinterpret_cast<_Float16*>(smem + OFF_KV);  // X; later Z^T
  _Float16* T1t = reinterpret_cast<_Float16*>(smem + OFF_T1);  // T1^T/T3^T; later W^T
  _Float16* T2t = reinterpret_cast<_Float16*>(smem + OFF_T2);
  float*    red = reinterpret_cast<float*>(smem + OFF_RED);

  const int tid  = threadIdx.x;
  const int lane = tid & 31;
  const int wave = tid >> 5;
  const int bh   = blockIdx.x;

  const float* gq = q + (size_t)bh * (N_TOK * DHEAD);
  const float* gk = k + (size_t)bh * (N_TOK * DHEAD);
  const float* gv = v + (size_t)bh * (N_TOK * DHEAD);

  // --- Stage 0: stream q,k -> LDS f16 row-major; v -> LDS f16 transposed ---
  {
    const float4* gq4 = reinterpret_cast<const float4*>(gq);
    const float4* gk4 = reinterpret_cast<const float4*>(gk);
    const float4* gv4 = reinterpret_cast<const float4*>(gv);
#pragma unroll 4
    for (int idx = tid; idx < (N_TOK * DHEAD) / 4; idx += 256) {
      const int base = idx * 4;
      const int row = base >> 6, col = base & 63;
      float4 xq = gq4[idx], xk = gk4[idx], xv = gv4[idx];
      v4h hq = {(_Float16)xq.x, (_Float16)xq.y, (_Float16)xq.z, (_Float16)xq.w};
      v4h hk = {(_Float16)xk.x, (_Float16)xk.y, (_Float16)xk.z, (_Float16)xk.w};
      *reinterpret_cast<v4h*>(&Qh[row * LDT + col]) = hq;
      *reinterpret_cast<v4h*>(&Kh[row * LDT + col]) = hk;
      VhT[(col + 0) * LD3 + row] = (_Float16)xv.x;
      VhT[(col + 1) * LD3 + row] = (_Float16)xv.y;
      VhT[(col + 2) * LD3 + row] = (_Float16)xv.z;
      VhT[(col + 3) * LD3 + row] = (_Float16)xv.w;
    }
  }
  __syncthreads();

  // --- Stage 1: landmark pooling (2x2 mean over 16x16 grid) from LDS f16 ---
  for (int idx = tid; idx < (LMK * DHEAD) / 4; idx += 256) {
    const int l = idx >> 4;           // landmark 0..63 (16 v4 chunks per row)
    const int d = (idx & 15) * 4;
    const int i = l >> 3, j = l & 7;
    const int r00 = (2 * i) * 16 + 2 * j;
    v4h q0 = *reinterpret_cast<const v4h*>(&Qh[(r00     ) * LDT + d]);
    v4h q1 = *reinterpret_cast<const v4h*>(&Qh[(r00 +  1) * LDT + d]);
    v4h q2 = *reinterpret_cast<const v4h*>(&Qh[(r00 + 16) * LDT + d]);
    v4h q3 = *reinterpret_cast<const v4h*>(&Qh[(r00 + 17) * LDT + d]);
    v4h k0 = *reinterpret_cast<const v4h*>(&Kh[(r00     ) * LDT + d]);
    v4h k1 = *reinterpret_cast<const v4h*>(&Kh[(r00 +  1) * LDT + d]);
    v4h k2 = *reinterpret_cast<const v4h*>(&Kh[(r00 + 16) * LDT + d]);
    v4h k3 = *reinterpret_cast<const v4h*>(&Kh[(r00 + 17) * LDT + d]);
    v4h oq, ok;
#pragma unroll
    for (int e = 0; e < 4; ++e) {
      oq[e] = (_Float16)(0.25f * ((float)q0[e] + (float)q1[e] + (float)q2[e] + (float)q3[e]));
      ok[e] = (_Float16)(0.25f * ((float)k0[e] + (float)k1[e] + (float)k2[e] + (float)k3[e]));
    }
    *reinterpret_cast<v4h*>(&QmH[l * LD64 + d]) = oq;
    *reinterpret_cast<v4h*>(&KmH[l * LD64 + d]) = ok;
  }
  __syncthreads();

  // --- Stage 2a: kernel_1 = softmax(q @ k_m^T) [256,64]; overwrite Qh ------
#pragma unroll
  for (int t = 0; t < 2; ++t) {
    const int mt = wave * 2 + t;
    v8f acc[4] = {{}, {}, {}, {}};
#pragma unroll
    for (int k0 = 0; k0 < 64; k0 += 32) {
      v16h a = load_A(Qh, LDT, mt * 16, k0, lane);
#pragma unroll
      for (int nt = 0; nt < 4; ++nt) {
        v16h b = load_Bt(KmH, LD64, nt * 16, k0, lane);
        acc[nt] = wmma16(a, b, acc[nt]);
      }
    }
    row_softmax<4>(acc, lane);
#pragma unroll
    for (int nt = 0; nt < 4; ++nt)
      store_tile_h(Qh, LDT, mt * 16, nt * 16, lane, acc[nt], 0.0f, 1.0f);
  }

  // --- Stage 2b: waves 0-3: kernel_3 = softmax(q_m @ k^T) [64,256]
  //               waves 4-7: kernel_2 = softmax(q_m @ k_m^T) [64,64]
  if (wave < 4) {
    const int mt = wave;
    v8f acc[16];
#pragma unroll
    for (int nt = 0; nt < 16; ++nt) acc[nt] = (v8f){};
#pragma unroll
    for (int k0 = 0; k0 < 64; k0 += 32) {
      v16h a = load_A(QmH, LD64, mt * 16, k0, lane);
#pragma unroll
      for (int nt = 0; nt < 16; ++nt) {
        v16h b = load_Bt(Kh, LDT, nt * 16, k0, lane);
        acc[nt] = wmma16(a, b, acc[nt]);
      }
    }
    row_softmax<16>(acc, lane);
#pragma unroll
    for (int nt = 0; nt < 16; ++nt)
      store_tile_h(K3h, LD3, mt * 16, nt * 16, lane, acc[nt], 0.0f, 1.0f);
  } else {
    const int mt = wave - 4;
    v8f acc[4] = {{}, {}, {}, {}};
#pragma unroll
    for (int k0 = 0; k0 < 64; k0 += 32) {
      v16h a = load_A(QmH, LD64, mt * 16, k0, lane);
#pragma unroll
      for (int nt = 0; nt < 4; ++nt) {
        v16h b = load_Bt(KmH, LD64, nt * 16, k0, lane);
        acc[nt] = wmma16(a, b, acc[nt]);
      }
    }
    row_softmax<4>(acc, lane);
#pragma unroll
    for (int nt = 0; nt < 4; ++nt)
      store_tile_h(K2h, LD64, mt * 16, nt * 16, lane, acc[nt], 0.0f, 1.0f);
  }
  __syncthreads();

  // --- Stage 3: NS init: V0 = K2^T / max_j(colsum_j(K2)), both layouts -----
  if (tid < 64) {
    float s = 0.0f;
    for (int i = 0; i < 64; ++i) s += (float)K2h[i * LD64 + tid];
    red[tid] = s;
  }
  __syncthreads();
  if (tid < 32) {
    float m = fmaxf(red[tid], red[tid + 32]);
#pragma unroll
    for (int off = 1; off < 32; off <<= 1) m = fmaxf(m, __shfl_xor(m, off, 32));
    if (tid == 0) red[64] = 1.0f / m;
  }
  __syncthreads();
  {
    const float invden = red[64];
    for (int idx = tid; idx < LMK * LMK; idx += 256) {
      const int i = idx >> 6, j = idx & 63;
      VaN[i * LD64 + j] = (_Float16)((float)K2h[j * LD64 + i] * invden);  // V0 = K2^T/d
      VaT[i * LD64 + j] = (_Float16)((float)K2h[i * LD64 + j] * invden);  // V0^T = K2/d
    }
  }
  __syncthreads();

  // --- Stage 4: Newton-Schulz: V <- 0.25 V (13I - X(15I - X(7I - X))) ------
  _Float16 *VcN = VaN, *VcT = VaT, *VnN = VbN, *VnT = VbT;
  for (int it = 0; it < 6; ++it) {
    // X = K2 @ V : store X (normal, A-use) and T1^T = (7I - X)^T
    mm64(K2h, LD64, VcT, LD64, KVh, 0.0f, 1.0f, T1t, 7.0f, -1.0f, wave, lane);
    __syncthreads();
    // T2^T = (15I - X @ T1)^T
    mm64(KVh, LD64, T1t, LD64, nullptr, 0.0f, 0.0f, T2t, 15.0f, -1.0f, wave, lane);
    __syncthreads();
    // T3^T = (13I - X @ T2)^T  (reuse T1t)
    mm64(KVh, LD64, T2t, LD64, nullptr, 0.0f, 0.0f, T1t, 13.0f, -1.0f, wave, lane);
    __syncthreads();
    // Vn = 0.25 V @ T3 : dual store (normal + transposed)
    mm64(VcN, LD64, T1t, LD64, VnN, 0.0f, 0.25f, VnT, 0.0f, 0.25f, wave, lane);
    __syncthreads();
    _Float16* t0 = VcN; VcN = VnN; VnN = t0;
    _Float16* t1 = VcT; VcT = VnT; VnT = t1;
  }
  // pinv(kernel_2) now in VcN / VcT.

  // --- Stage 5: Z = kernel_3 @ v  [64,256]@[256,64] -> Z^T in KVh ----------
#pragma unroll
  for (int t = 0; t < 2; ++t) {
    const int tile = wave * 2 + t;
    const int mt = tile >> 2, nt = tile & 3;
    v8f acc = {};
#pragma unroll
    for (int k0 = 0; k0 < 256; k0 += 32) {
      v16h a = load_A(K3h, LD3, mt * 16, k0, lane);
      v16h b = load_Bt(VhT, LD3, nt * 16, k0, lane);
      acc = wmma16(a, b, acc);
    }
    store_tile_hT(KVh, LD64, mt * 16, nt * 16, lane, acc, 0.0f, 1.0f);
  }
  __syncthreads();

  // --- Stage 6: W = pinv @ Z -> W^T in T1t ---------------------------------
  mm64(VcN, LD64, KVh, LD64, nullptr, 0.0f, 0.0f, T1t, 0.0f, 1.0f, wave, lane);
  __syncthreads();

  // --- Stage 7: out^T = W^T @ k1^T  ->  vectorized b128 global stores ------
  // A = W^T (T1t, row-major); B(k,n) = k1[n][k] via load_Bt on row-major Qh.
  // acc element r maps to out[token][d0+r] with 8 consecutive d -> 2x float4.
  float* outp = out + (size_t)bh * (N_TOK * DHEAD);
#pragma unroll
  for (int t = 0; t < 8; ++t) {
    const int tile = wave * 8 + t;   // 64 tiles: 4 d-tiles x 16 token-tiles
    const int mt = tile >> 4;        // d tile 0..3
    const int nt = tile & 15;        // token tile 0..15
    v8f acc = {};
#pragma unroll
    for (int k0 = 0; k0 < 64; k0 += 32) {
      v16h a = load_A(T1t, LD64, mt * 16, k0, lane);   // W^T
      v16h b = load_Bt(Qh, LDT, nt * 16, k0, lane);    // k1 rows = tokens
      acc = wmma16(a, b, acc);
    }
    const int tok = nt * 16 + (lane & 15);
    const int d0  = mt * 16 + ((lane & 16) ? 8 : 0);
    float4 w0 = {acc[0], acc[1], acc[2], acc[3]};
    float4 w1 = {acc[4], acc[5], acc[6], acc[7]};
    float4* p = reinterpret_cast<float4*>(&outp[(size_t)tok * DHEAD + d0]);
    p[0] = w0;
    p[1] = w1;
  }
}

// ---------------------------------------------------------------------------
extern "C" void kernel_launch(void* const* d_in, const int* in_sizes, int n_in,
                              void* d_out, int out_size, void* d_ws, size_t ws_size,
                              hipStream_t stream) {
  (void)in_sizes; (void)n_in; (void)out_size; (void)d_ws; (void)ws_size;
  const float* q = (const float*)d_in[0];
  const float* k = (const float*)d_in[1];
  const float* v = (const float*)d_in[2];
  float* out = (float*)d_out;

  hipFuncSetAttribute(reinterpret_cast<const void*>(nystrom_attn_kernel),
                      hipFuncAttributeMaxDynamicSharedMemorySize, (int)SMEM_BYTES);
  nystrom_attn_kernel<<<dim3(BH_TOTAL), dim3(256), SMEM_BYTES, stream>>>(q, k, v, out);
}
